// MethodGCN_38912403702117
// MI455X (gfx1250) — matrix operand order repaired and checked by reading, not compiled, not run on hardware
//
#include <hip/hip_runtime.h>
#include <hip/hip_bf16.h>
#include <math.h>

typedef __attribute__((ext_vector_type(16))) _Float16 v16h;
typedef __attribute__((ext_vector_type(8)))  _Float16 v8h;
typedef __attribute__((ext_vector_type(4)))  _Float16 v4h;
typedef __attribute__((ext_vector_type(8)))  float    v8f;
typedef __attribute__((ext_vector_type(4)))  float    v4f;

// ---------------------------------------------------------------------------
// Pre-convert weights: Wt[n][k] (f16, [Npad x Kpad], zero-padded) from
// W[k][n] (f32, row-major K x N). Runs once per layer, tiny.
// ---------------------------------------------------------------------------
__global__ __launch_bounds__(256) void gcn_wt_convert(
    const float* __restrict__ W, _Float16* __restrict__ Wt,
    int K, int N, int Kpad, int Npad)
{
    int i = blockIdx.x * 256 + threadIdx.x;
    if (i >= Npad * Kpad) return;
    int n = i / Kpad, k = i - n * Kpad;
    float v = (n < N && k < K) ? W[(size_t)k * N + n] : 0.f;
    Wt[i] = (_Float16)v;
}

// ---------------------------------------------------------------------------
// GEMM: C[M,N] = A[M,K] (fp32, stride lda) @ W (as pre-transposed f16 Wt[n][k])
// Core: v_wmma_f32_16x16x32_f16, fp32 accum/out (stride ldc).
// Block: 256 threads = 8 waves (wave32). BM=128, BN = NT*16, KS=32.
// ---------------------------------------------------------------------------
template<int NT>
__global__ __launch_bounds__(256) void gcn_gemm_wmma_f16(
    const float* __restrict__ A, const _Float16* __restrict__ Wt,
    float* __restrict__ C, int M, int N, int K, int lda, int ldc, int kpad)
{
    constexpr int BM = 128, KS = 32, BN = NT * 16;
    constexpr int AS = KS + 8;   // LDS row stride in f16 (80B: 16B-aligned rows, skewed banks)
    constexpr int BS = KS + 8;

    __shared__ alignas(16) _Float16 la[BM * AS];
    __shared__ alignas(16) _Float16 lb[BN * BS];

    const int tid  = threadIdx.x;
    const int lane = tid & 31;
    const int wave = tid >> 5;
    const int m0   = blockIdx.x * BM;
    const bool mfull = (m0 + BM <= M);
    const int Kpad = (K + KS - 1) & ~(KS - 1);

    v8f acc[NT];
    const v8f vzero = {0.f, 0.f, 0.f, 0.f, 0.f, 0.f, 0.f, 0.f};
#pragma unroll
    for (int t = 0; t < NT; ++t) acc[t] = vzero;

    // WMMA fragment addressing (wave32, 16x16x32 f16):
    // A: lane -> row mA = wave*16 + lane%16; K chunks {kh..kh+7},{kh+16..kh+23}, kh=(lane/16)*8
    // B: lane -> col  n = lane%16; K chunks {kh..kh+7},{kh+16..kh+23} (lb stores [n][k])
    const int mA  = (wave << 4) + (lane & 15);
    const int kh  = (lane >> 4) << 3;
    const int kh2 = (lane >> 4) << 4;

    for (int k0 = 0; k0 < Kpad; k0 += KS) {
        const bool ktail = (k0 + KS > K);
        // ---- stage A tile (BM x KS), fp32 -> f16 ----
        if (mfull && !ktail) {
            // Fast path: 4x 16B loads/thread (align-4 safe via memcpy), ds_store_b64.
#pragma unroll
            for (int i = 0; i < 4; ++i) {
                int idx = tid + (i << 8);           // 0..1023 float4-chunks
                int r   = idx >> 3;
                int c4  = (idx & 7) << 2;
                v4f v;
                __builtin_memcpy(&v, &A[(size_t)(m0 + r) * lda + (k0 + c4)], 16);
                v4h h = { (_Float16)v.x, (_Float16)v.y, (_Float16)v.z, (_Float16)v.w };
                *(v4h*)&la[r * AS + c4] = h;
            }
            // prefetch next A K-block (global_prefetch_b8)
            if (k0 + KS < Kpad) {
                int r = tid >> 1;
                __builtin_prefetch(&A[(size_t)(m0 + r) * lda + k0 + KS + ((tid & 1) << 4)], 0, 1);
            }
        } else {
            // Edge path (partial M tile and/or ragged K tail): guarded scalar.
            for (int i = tid; i < BM * KS; i += 256) {
                int r = i >> 5, c = i & (KS - 1);
                int gm = m0 + r, gk = k0 + c;
                float v = (gm < M && gk < K) ? A[(size_t)gm * lda + gk] : 0.f;
                la[r * AS + c] = (_Float16)v;
            }
        }
        // ---- stage B tile: straight 16B f16 copy from pre-padded Wt ----
        {
            constexpr int BCH = (BN * KS) / 8;      // 16B chunks
            for (int i = tid; i < BCH; i += 256) {
                int n  = i >> 2;                    // KS/8 = 4 chunks per row
                int c8 = (i & 3) << 3;
                *(v8h*)&lb[n * BS + c8] =
                    *(const v8h*)&Wt[(size_t)n * kpad + k0 + c8];
            }
        }
        __syncthreads();

        // ---- fragments: ds_load_b128 pairs, then WMMA ----
        v8h alo = *(const v8h*)&la[mA * AS + kh];
        v8h ahi = *(const v8h*)&la[mA * AS + kh + 16];
        v16h af = __builtin_shufflevector(alo, ahi,
                    0,1,2,3,4,5,6,7,8,9,10,11,12,13,14,15);
#pragma unroll
        for (int t = 0; t < NT; ++t) {
            int nb = (t << 4) + (lane & 15);
            v8h blo = *(const v8h*)&lb[nb * BS + kh];
            v8h bhi = *(const v8h*)&lb[nb * BS + kh + 16];
            v16h bf = __builtin_shufflevector(blo, bhi,
                        0,1,2,3,4,5,6,7,8,9,10,11,12,13,14,15);
            acc[t] = __builtin_amdgcn_wmma_f32_16x16x32_f16(
                         false, af, false, bf, (short)0, acc[t], false, false);
        }
        __syncthreads();
    }

    // ---- store: C/D layout — VGPR r: M = r + 8*(lane/16), N = lane%16 ----
    const int mloc = (wave << 4) + ((lane >> 4) << 3);
    const int ncol = lane & 15;
#pragma unroll
    for (int t = 0; t < NT; ++t) {
        int n = (t << 4) + ncol;
        if (n < N) {
#pragma unroll
            for (int r = 0; r < 8; ++r) {
                int m = m0 + mloc + r;
                if (m < M) C[(size_t)m * ldc + n] = acc[t][r];
            }
        }
    }
}

// ---------------------------------------------------------------------------
// SpMM scatter: out[dst[e], :] += val[e] * sup[src[e], :]
// FV = row width in float4. Support fits in 192MB L2 -> gathers + fp32
// global_atomic_add_f32 run at L2 bandwidth.
// ---------------------------------------------------------------------------
template<int FV>
__global__ __launch_bounds__(256) void gcn_spmm_atomic(
    const int* __restrict__ esrc, const int* __restrict__ edst,
    const float* __restrict__ eval, const float* __restrict__ sup,
    float* __restrict__ out, int E)
{
    long long gid = (long long)blockIdx.x * 256 + threadIdx.x;
    int e = (int)(gid / FV);
    int q = (int)(gid % FV);
    if (e >= E) return;
    int s = esrc[e], d = edst[e];
    float v = eval[e];
    const float4 m = ((const float4*)(sup + (size_t)s * (FV * 4)))[q];
    float* o = out + (size_t)d * (FV * 4) + q * 4;
    __hip_atomic_fetch_add(o + 0, m.x * v, __ATOMIC_RELAXED, __HIP_MEMORY_SCOPE_AGENT);
    __hip_atomic_fetch_add(o + 1, m.y * v, __ATOMIC_RELAXED, __HIP_MEMORY_SCOPE_AGENT);
    __hip_atomic_fetch_add(o + 2, m.z * v, __ATOMIC_RELAXED, __HIP_MEMORY_SCOPE_AGENT);
    __hip_atomic_fetch_add(o + 3, m.w * v, __ATOMIC_RELAXED, __HIP_MEMORY_SCOPE_AGENT);
}

// Seed accumulator buffer with bias (cols >= F get 0) so atomics land on b + sum.
__global__ __launch_bounds__(256) void gcn_init_bias(
    float* __restrict__ out, const float* __restrict__ b,
    int F, int stride, long long total)
{
    long long i = (long long)blockIdx.x * 256 + threadIdx.x;
    if (i >= total) return;
    int c = (int)(i & (stride - 1));
    out[i] = (c < F) ? b[c] : 0.f;
}

__global__ __launch_bounds__(256) void gcn_relu(float* __restrict__ x, long long n)
{
    long long i = (long long)blockIdx.x * 256 + threadIdx.x;
    if (i < n) x[i] = fmaxf(x[i], 0.f);
}

__global__ __launch_bounds__(256) void gcn_log_softmax7(
    const float* __restrict__ h, float* __restrict__ out, int M)
{
    int i = blockIdx.x * 256 + threadIdx.x;
    if (i >= M) return;
    const float* r = h + (size_t)i * 8;
    float v[7];
    float mx = -INFINITY;
#pragma unroll
    for (int j = 0; j < 7; ++j) { v[j] = r[j]; mx = fmaxf(mx, v[j]); }
    float s = 0.f;
#pragma unroll
    for (int j = 0; j < 7; ++j) s += __expf(v[j] - mx);
    float l = mx + __logf(s);
    float* o = out + (size_t)i * 7;
#pragma unroll
    for (int j = 0; j < 7; ++j) o[j] = v[j] - l;
}

// ---------------------------------------------------------------------------
extern "C" void kernel_launch(void* const* d_in, const int* in_sizes, int n_in,
                              void* d_out, int out_size, void* d_ws, size_t ws_size,
                              hipStream_t stream) {
    const float* x    = (const float*)d_in[0];
    const int*   esrc = (const int*)d_in[1];
    const int*   edst = (const int*)d_in[2];
    const float* evl  = (const float*)d_in[3];
    const float* W1   = (const float*)d_in[4];
    const float* b1   = (const float*)d_in[5];
    const float* W2   = (const float*)d_in[6];
    const float* b2   = (const float*)d_in[7];
    const float* W3   = (const float*)d_in[8];
    const float* b3   = (const float*)d_in[9];

    const int M = in_sizes[0] / 1433;   // 100000
    const int E = in_sizes[1];          // 3200000

    // Workspace (floats); activation buffers ping-pong: [0, M*128) floats.
    float* ws   = (float*)d_ws;
    float* sup1 = ws;                                   // [M,64]
    float* h1   = ws + (size_t)M * 64;                  // [M,64]
    float* sup2 = ws;                                   // [M,32]  (sup1 dead)
    float* h2   = ws + (size_t)M * 32;                  // [M,32]
    float* sup3 = ws + (size_t)M * 64;                  // [M,8]   (h1 dead)
    float* h3   = ws + (size_t)M * 72;                  // [M,8]
    // Pre-converted f16 transposed weights after the fp32 region.
    _Float16* W1t = (_Float16*)(ws + (size_t)M * 128);  // [64 x 1440]
    _Float16* W2t = W1t + 64 * 1440;                    // [32 x 64]
    _Float16* W3t = W2t + 32 * 64;                      // [16 x 32]

    const int gemm_blocks = (M + 127) / 128;

    // ---- one-time (per launch) weight transpose+convert ----
    gcn_wt_convert<<<(64 * 1440 + 255) / 256, 256, 0, stream>>>(W1, W1t, 1433, 60, 1440, 64);
    gcn_wt_convert<<<(32 * 64   + 255) / 256, 256, 0, stream>>>(W2, W2t, 60, 30, 64, 32);
    gcn_wt_convert<<<(16 * 32   + 255) / 256, 256, 0, stream>>>(W3, W3t, 30, 7, 32, 16);

    // ---- Layer 1: support = x @ W1 ; h1 = relu(A@support + b1) ----
    gcn_gemm_wmma_f16<4><<<gemm_blocks, 256, 0, stream>>>(x, W1t, sup1, M, 60, 1433, 1433, 64, 1440);
    {
        long long tot = (long long)M * 64;
        gcn_init_bias<<<(int)((tot + 255) / 256), 256, 0, stream>>>(h1, b1, 60, 64, tot);
        long long thr = (long long)E * 16;
        gcn_spmm_atomic<16><<<(int)((thr + 255) / 256), 256, 0, stream>>>(esrc, edst, evl, sup1, h1, E);
        gcn_relu<<<(int)((tot + 255) / 256), 256, 0, stream>>>(h1, tot);
    }

    // ---- Layer 2: support = h1 @ W2 ; h2 = A@support + b2 ----
    gcn_gemm_wmma_f16<2><<<gemm_blocks, 256, 0, stream>>>(h1, W2t, sup2, M, 30, 60, 64, 32, 64);
    {
        long long tot = (long long)M * 32;
        gcn_init_bias<<<(int)((tot + 255) / 256), 256, 0, stream>>>(h2, b2, 30, 32, tot);
        long long thr = (long long)E * 8;
        gcn_spmm_atomic<8><<<(int)((thr + 255) / 256), 256, 0, stream>>>(esrc, edst, evl, sup2, h2, E);
    }

    // ---- Layer 3: support = h2 @ W3 ; h3 = A@support + b3 ; log_softmax ----
    gcn_gemm_wmma_f16<1><<<gemm_blocks, 256, 0, stream>>>(h2, W3t, sup3, M, 7, 30, 32, 8, 32);
    {
        long long tot = (long long)M * 8;
        gcn_init_bias<<<(int)((tot + 255) / 256), 256, 0, stream>>>(h3, b3, 7, 8, tot);
        long long thr = (long long)E * 2;
        gcn_spmm_atomic<2><<<(int)((thr + 255) / 256), 256, 0, stream>>>(esrc, edst, evl, sup3, h3, E);
    }
    gcn_log_softmax7<<<(M + 255) / 256, 256, 0, stream>>>(h3, (float*)d_out, M);
}
